// PatchGCN_53601191854260
// MI455X (gfx1250) — matrix operand order repaired and checked by reading, not compiled, not run on hardware
//
#include <hip/hip_runtime.h>
#include <hip/hip_bf16.h>
#include <math.h>

// ---------------------------------------------------------------------------
// PatchGCN forward for MI455X (gfx1250, wave32).
// Dense GEMMs (134 GFLOP) on v_wmma_f32_16x16x32_bf16. Weights pre-converted
// to bf16 [N][K]; activations carried as bf16 shadows (xcat_bf is 61 MB ->
// L2-resident across the 8 N-blocks of phi/attn). B tiles staged into LDS by
// the Tensor Data Mover (tensor_load_to_lds + s_wait_tensorcnt) with 16B row
// padding for conflict-free ds_load_b128 fragment reads; manual fallback if
// the builtin is unavailable. Edge softmax agg = uint atomicMax + atomicAdd.
// ---------------------------------------------------------------------------

#define N_NODES 60000
#define N_EDGES 480000
#define D_IN    1024
#define HID     128
#define CAT     512
#define CLIN    32
#define WSI_H   256
#define CONCAT_H 64
#define EPS_MSG 1e-7f
#define LN_EPS  1e-5f

typedef __attribute__((ext_vector_type(16))) __bf16          v16bf;
typedef __attribute__((ext_vector_type(16))) unsigned short  v16us;
typedef __attribute__((ext_vector_type(8)))  unsigned short  v8us;
typedef __attribute__((ext_vector_type(8)))  float           v8f;
typedef __attribute__((ext_vector_type(4)))  unsigned int    u32x4;
typedef __attribute__((ext_vector_type(8)))  int             i32x8;
typedef __attribute__((ext_vector_type(4)))  int             i32x4;

#if __has_builtin(__builtin_amdgcn_tensor_load_to_lds) && \
    __has_builtin(__builtin_amdgcn_s_wait_tensorcnt)
#define USE_TDM 1
#else
#define USE_TDM 0
#endif

// ---- input index map (top-level dict in insertion order; params pytree
// ---- flattened with sorted dict keys, lists in order) ----------------------
enum {
  IN_X = 0, IN_EDGE = 1, IN_CLIN = 2,
  P_ATTN_A_B = 3, P_ATTN_A_W = 4, P_ATTN_B_B = 5, P_ATTN_B_W = 6,
  P_ATTN_C_B = 7, P_ATTN_C_W = 8,
  P_CONV0 = 9,   // + l*7 ; within conv: lin1.b, lin1.w, lin2.b, lin2.w, ln.b, ln.g, t
  P_FC_B = 30, P_FC_W = 31,
  P_NORM0_B = 32, P_NORM0_G = 33, P_NORM1_B = 34, P_NORM1_G = 35,
  P_OUT1_B = 36, P_OUT1_W = 37, P_OUT2_B = 38, P_OUT2_W = 39,
  P_PHI_B = 40, P_PHI_W = 41, P_RHO_B = 42, P_RHO_W = 43,
  P_WSI_B = 44, P_WSI_W = 45
};
#define CV_LIN1_B 0
#define CV_LIN1_W 1
#define CV_LIN2_B 2
#define CV_LIN2_W 3
#define CV_LN_B   4
#define CV_LN_G   5
#define CV_T      6

// ---------------------------------------------------------------------------
// helpers
// ---------------------------------------------------------------------------
__device__ __forceinline__ unsigned short f32_to_bf16_bits(float f) {
  unsigned int u = __float_as_uint(f);
  u += 0x7FFFu + ((u >> 16) & 1u);       // round-to-nearest-even
  return (unsigned short)(u >> 16);
}

template <int ACT> __device__ __forceinline__ float apply_act(float v) {
  if (ACT == 1) return fmaxf(v, 0.0f);
  return v;
}

// ---------------------------------------------------------------------------
// Weight prep: W[K,N] fp32 -> Wt[N,K] bf16 (done once; tiles become K-runs)
// ---------------------------------------------------------------------------
__global__ __launch_bounds__(256)
void k_wprep(const float* __restrict__ W, unsigned short* __restrict__ Wt,
             int K, int N)
{
  int i = blockIdx.x * 256 + threadIdx.x;
  if (i >= K * N) return;
  int k = i / N, n = i % N;
  Wt[(size_t)n * K + k] = f32_to_bf16_bits(W[i]);
}

// ---------------------------------------------------------------------------
// Tiled bf16 WMMA GEMM:  C[M,N] = act(A[M,K] @ Wt[N,K]^T + bias[N])
// Block = 256 threads (8 waves), tile 64x64, K-step 32, double buffered.
// Wave w -> m-tile (w&3), n-tiles 2*(w>>2), 2*(w>>2)+1.
// A staged row-major stride 40 (ds_store_b128/ds_load_b128); B staged by TDM
// (row pad 16B -> stride 40) or manual v8us copies.
// ---------------------------------------------------------------------------
#define TM 64
#define TN 64
#define TK 32
#define SA_LD  40   // ushorts per A row
#define SBT_LD 40   // ushorts per B row (TK=32 data + 8 pad; TDM pad_amount)

#if USE_TDM
// 2D TDM descriptor: tile 64 rows x 32 bf16, row stride K elements, dest LDS
// padded by 16B every 64B -> LDS row stride 80B = SBT_LD ushorts.
__device__ __forceinline__ void tdm_issue_b(const unsigned short* g,
                                            unsigned lds_off, int K, int N)
{
  unsigned long long ga = (unsigned long long)(uintptr_t)g;
  u32x4 g0;
  g0[0] = 1u;                                                  // count=1
  g0[1] = lds_off;                                             // lds_addr
  g0[2] = (unsigned)ga;                                        // global lo
  g0[3] = ((unsigned)(ga >> 32) & 0x01FFFFFFu) | 0x80000000u;  // hi | type=2
  i32x8 g1;
  //      data_size=2B     pad_en     interval=64B  amount=16B
  g1[0] = (int)((1u << 16) | (1u << 20) | (3u << 22) | (3u << 25));
  g1[1] = (int)(((unsigned)K & 0xFFFFu) << 16);                // tdim0 lo
  g1[2] = (int)((((unsigned)K >> 16) & 0xFFFFu) |
                (((unsigned)N & 0xFFFFu) << 16));              // tdim0 hi|tdim1 lo
  g1[3] = (int)((((unsigned)N >> 16) & 0xFFFFu) | (32u << 16)); // tdim1 hi|tile0=32
  g1[4] = 64;                                                  // tile1=64, tile2=0
  g1[5] = K;                                                   // dim0_stride lo
  g1[6] = 0;
  g1[7] = 0;
  i32x4 z4 = {};
  i32x8 z8 = {};                       // groups 2/3 unused (2D tensor)
  __builtin_amdgcn_tensor_load_to_lds(g0, g1, z4, z4, z8, 0);
}
#endif

template <int ACT, int ABF>
__global__ __launch_bounds__(256)
void k_gemm_wmma(const void* __restrict__ Aptr, int lda,
                 const unsigned short* __restrict__ Bt,   // [N][K] bf16
                 const float* __restrict__ bias,
                 float* __restrict__ C, int ldc,
                 unsigned short* __restrict__ Cbf,        // optional bf16 shadow
                 int M, int N, int K)
{
  __shared__ __align__(16) unsigned short sA [2][TM * SA_LD];
  __shared__ __align__(16) unsigned short sB [2][TN * SBT_LD];

  const float*          Af = (const float*)Aptr;
  const unsigned short* Ab = (const unsigned short*)Aptr;

  const int tid  = threadIdx.x;
  const int lane = tid & 31;
  const int wave = tid >> 5;
  const int half = lane >> 4;
  const int r    = lane & 15;
  const int mt   = wave & 3;
  const int nt0  = (wave >> 2) * 2;

  const int blockM = blockIdx.x * TM;
  const int blockN = blockIdx.y * TN;

  v8f acc0 = {};
  v8f acc1 = {};

  // staging roles
  const int arow = tid >> 2;            // 0..63
  const int acol = (tid & 3) * 8;       // 0,8,16,24
  const int brow = tid >> 2;            // 0..63 (B fallback row)
  const int bseg = (tid & 3) * 8;       // 0..24

  const int  agrow  = blockM + arow;
  const bool avalid = (agrow < M);
  const int  asafe  = avalid ? agrow : (M - 1);
  const float amask = avalid ? 1.0f : 0.0f;

  float4 av0, av1;                      // fp32-A staging
  v8us   abf;                           // bf16-A staging
  v8us   bst;                           // B fallback staging

  auto loadG = [&](int k0) {
    if (ABF) {
      abf = *(const v8us*)(Ab + (size_t)asafe * lda + k0 + acol);
    } else {
      const float* ap = Af + (size_t)asafe * lda + k0 + acol;
      av0 = *(const float4*)ap;
      av1 = *(const float4*)(ap + 4);
    }
#if !USE_TDM
    bst = *(const v8us*)(Bt + (size_t)(blockN + brow) * K + k0 + bseg);
#endif
  };

  auto storeL = [&](int buf) {
    if (ABF) {
      v8us z = {};
      *(v8us*)&sA[buf][arow * SA_LD + acol] = avalid ? abf : z;
    } else {
      v8us a8;
      a8[0] = f32_to_bf16_bits(av0.x * amask);
      a8[1] = f32_to_bf16_bits(av0.y * amask);
      a8[2] = f32_to_bf16_bits(av0.z * amask);
      a8[3] = f32_to_bf16_bits(av0.w * amask);
      a8[4] = f32_to_bf16_bits(av1.x * amask);
      a8[5] = f32_to_bf16_bits(av1.y * amask);
      a8[6] = f32_to_bf16_bits(av1.z * amask);
      a8[7] = f32_to_bf16_bits(av1.w * amask);
      *(v8us*)&sA[buf][arow * SA_LD + acol] = a8;
    }
#if !USE_TDM
    *(v8us*)&sB[buf][brow * SBT_LD + bseg] = bst;
#endif
  };

  const int nsteps = K / TK;

#if USE_TDM
  const unsigned sb_lds0 = (unsigned)(uintptr_t)(const void*)&sB[0][0];
  if (wave == 0)
    tdm_issue_b(Bt + (size_t)blockN * K, sb_lds0, K, N);
#endif
  loadG(0);
  storeL(0);
#if USE_TDM
  if (wave == 0) __builtin_amdgcn_s_wait_tensorcnt(0);
#endif
  __syncthreads();

  int buf = 0;
  for (int s = 0; s < nsteps; ++s) {
    const bool more = (s + 1 < nsteps);
    if (more) {
#if USE_TDM
      if (wave == 0)
        tdm_issue_b(Bt + (size_t)blockN * K + (s + 1) * TK,
                    sb_lds0 + (unsigned)((buf ^ 1) * TN * SBT_LD * 2), K, N);
#endif
      loadG((s + 1) * TK);
      if (s + 2 < nsteps) {                    // keep L2 warm
        if (ABF)
          __builtin_prefetch(Ab + (size_t)asafe * lda + (s + 2) * TK + acol, 0, 0);
        else
          __builtin_prefetch(Af + (size_t)asafe * lda + (s + 2) * TK + acol, 0, 0);
      }
    }

    // ---- fragments: contiguous 16B runs -> ds_load_b128 -------------------
    const unsigned short* pa = &sA[buf][(mt * 16 + r) * SA_LD];
    v8us alo = *(const v8us*)(pa + 16 * half);         // K 16h..16h+7
    v8us ahi = *(const v8us*)(pa + 16 * half + 8);     // K 16h+8..16h+15
    v16bf af = __builtin_bit_cast(v16bf,
        __builtin_shufflevector(alo, ahi, 0,1,2,3,4,5,6,7,8,9,10,11,12,13,14,15));

    const unsigned short* pb0 = &sB[buf][(nt0 * 16 + r) * SBT_LD + 16 * half];
    v8us b0lo = *(const v8us*)pb0;
    v8us b0hi = *(const v8us*)(pb0 + 8);
    v16bf bf0 = __builtin_bit_cast(v16bf,
        __builtin_shufflevector(b0lo, b0hi, 0,1,2,3,4,5,6,7,8,9,10,11,12,13,14,15));

    const unsigned short* pb1 = &sB[buf][((nt0 + 1) * 16 + r) * SBT_LD + 16 * half];
    v8us b1lo = *(const v8us*)pb1;
    v8us b1hi = *(const v8us*)(pb1 + 8);
    v16bf bf1 = __builtin_bit_cast(v16bf,
        __builtin_shufflevector(b1lo, b1hi, 0,1,2,3,4,5,6,7,8,9,10,11,12,13,14,15));

    acc0 = __builtin_amdgcn_wmma_f32_16x16x32_bf16(false, af, false, bf0,
                                                   (short)0, acc0, false, false);
    acc1 = __builtin_amdgcn_wmma_f32_16x16x32_bf16(false, af, false, bf1,
                                                   (short)0, acc1, false, false);

    if (more) {
      storeL(buf ^ 1);
#if USE_TDM
      if (wave == 0) __builtin_amdgcn_s_wait_tensorcnt(0);
#endif
    }
    __syncthreads();
    buf ^= 1;
  }

  // epilogue: VGPR i -> row mt*16 + 8*half + i, col nt*16 + r
  #pragma unroll
  for (int i = 0; i < 8; ++i) {
    int row = blockM + mt * 16 + half * 8 + i;
    if (row < M) {
      int c0 = blockN + nt0 * 16 + r;
      int c1 = c0 + 16;
      float v0 = apply_act<ACT>(acc0[i] + bias[c0]);
      float v1 = apply_act<ACT>(acc1[i] + bias[c1]);
      C[(size_t)row * ldc + c0] = v0;
      C[(size_t)row * ldc + c1] = v1;
      if (Cbf) {
        Cbf[(size_t)row * ldc + c0] = f32_to_bf16_bits(v0);
        Cbf[(size_t)row * ldc + c1] = f32_to_bf16_bits(v1);
      }
    }
  }
}

// ---------------------------------------------------------------------------
// Edge softmax aggregation (GENConv, t>0 so all scores positive -> uint max)
// ---------------------------------------------------------------------------
__global__ __launch_bounds__(256)
void k_edge_max(const int* __restrict__ src, const int* __restrict__ dst,
                const float* __restrict__ h, int ldh,
                unsigned int* __restrict__ mx, const float* __restrict__ tptr,
                int E)
{
  int e = blockIdx.x * 2 + (threadIdx.x >> 7);
  int c = threadIdx.x & 127;
  if (e >= E) return;
  int s = src[e], d = dst[e];
  float m = fmaxf(h[(size_t)s * ldh + c], 0.0f) + EPS_MSG;
  float a = m * tptr[0];
  atomicMax(&mx[(size_t)d * HID + c], __float_as_uint(a));
}

__global__ __launch_bounds__(256)
void k_edge_acc(const int* __restrict__ src, const int* __restrict__ dst,
                const float* __restrict__ h, int ldh,
                const unsigned int* __restrict__ mx,
                float* __restrict__ num, float* __restrict__ den,
                const float* __restrict__ tptr, int E)
{
  int e = blockIdx.x * 2 + (threadIdx.x >> 7);
  int c = threadIdx.x & 127;
  if (e >= E) return;
  int s = src[e], d = dst[e];
  float m = fmaxf(h[(size_t)s * ldh + c], 0.0f) + EPS_MSG;
  float a = m * tptr[0];
  size_t idx = (size_t)d * HID + c;
  float ev = __expf(a - __uint_as_float(mx[idx]));
  atomicAdd(&den[idx], ev);
  atomicAdd(&num[idx], m * ev);
}

// node_bf = bf16( agg + hin )  (feeds lin1 GEMM directly)
__global__ __launch_bounds__(256)
void k_combine(const float* __restrict__ num, const float* __restrict__ den,
               const float* __restrict__ hin, int ldh,
               unsigned short* __restrict__ node_bf)
{
  int i = blockIdx.x * 256 + threadIdx.x;
  if (i >= N_NODES * HID) return;
  int n = i >> 7, c = i & 127;
  float d = den[i];
  float agg = (d > 0.0f) ? num[i] / d : 0.0f;          // empty segment -> 0
  node_bf[i] = f32_to_bf16_bits(agg + hin[(size_t)n * ldh + c]);
}

// ---------------------------------------------------------------------------
// Wave-per-row LayerNorm + ReLU (+ optional residual): Y = res + relu(LN(X))
// Optional bf16 shadow output for downstream WMMA consumers.
// ---------------------------------------------------------------------------
__global__ __launch_bounds__(256)
void k_layernorm_relu(const float* __restrict__ X, int ldx, int D,
                      const float* __restrict__ g, const float* __restrict__ b,
                      const float* __restrict__ res, int ldr,
                      float* __restrict__ Y, unsigned short* __restrict__ Ybf,
                      int ldy, int N)
{
  int wave = threadIdx.x >> 5, lane = threadIdx.x & 31;
  int n = blockIdx.x * 8 + wave;
  if (n >= N) return;
  const float* xp = X + (size_t)n * ldx;
  float s = 0.0f, ss = 0.0f;
  for (int c = lane; c < D; c += 32) { float v = xp[c]; s += v; ss += v * v; }
  #pragma unroll
  for (int m = 16; m; m >>= 1) { s += __shfl_xor(s, m, 32); ss += __shfl_xor(ss, m, 32); }
  float mean = s / (float)D;
  float var  = ss / (float)D - mean * mean;
  float rstd = rsqrtf(var + LN_EPS);
  for (int c = lane; c < D; c += 32) {
    float v = fmaxf((xp[c] - mean) * rstd * g[c] + b[c], 0.0f);
    if (res) v += res[(size_t)n * ldr + c];
    if (Y)   Y  [(size_t)n * ldy + c] = v;
    if (Ybf) Ybf[(size_t)n * ldy + c] = f32_to_bf16_bits(v);
  }
}

// ---------------------------------------------------------------------------
// Attention scores: A[n] = sum_c tanh(a)*sigmoid(b)*wc[c] + cb
// ---------------------------------------------------------------------------
__global__ __launch_bounds__(256)
void k_attn_score(const float* __restrict__ a, const float* __restrict__ b,
                  const float* __restrict__ cw, const float* __restrict__ cb,
                  float* __restrict__ A, int N)
{
  int wave = threadIdx.x >> 5, lane = threadIdx.x & 31;
  int n = blockIdx.x * 8 + wave;
  if (n >= N) return;
  const float* ap = a + (size_t)n * CAT;
  const float* bp = b + (size_t)n * CAT;
  float acc = 0.0f;
  for (int c = lane; c < CAT; c += 32) {
    float av = tanhf(ap[c]);
    float bv = 1.0f / (1.0f + __expf(-bp[c]));
    acc += av * bv * cw[c];
  }
  #pragma unroll
  for (int m = 16; m; m >>= 1) acc += __shfl_xor(acc, m, 32);
  if (lane == 0) A[n] = acc + cb[0];
}

// ---------------------------------------------------------------------------
// Softmax-over-nodes reductions
// ---------------------------------------------------------------------------
__global__ __launch_bounds__(256)
void k_max_partial(const float* __restrict__ A, int N, float* __restrict__ part)
{
  __shared__ float sm[256];
  float m = -3.4e38f;
  for (int i = blockIdx.x * 256 + threadIdx.x; i < N; i += gridDim.x * 256)
    m = fmaxf(m, A[i]);
  sm[threadIdx.x] = m; __syncthreads();
  for (int s = 128; s; s >>= 1) {
    if (threadIdx.x < s) sm[threadIdx.x] = fmaxf(sm[threadIdx.x], sm[threadIdx.x + s]);
    __syncthreads();
  }
  if (threadIdx.x == 0) part[blockIdx.x] = sm[0];
}

__global__ __launch_bounds__(256)
void k_max_final(const float* __restrict__ part, int G, float* __restrict__ sc)
{
  __shared__ float sm[256];
  float m = -3.4e38f;
  for (int i = threadIdx.x; i < G; i += 256) m = fmaxf(m, part[i]);
  sm[threadIdx.x] = m; __syncthreads();
  for (int s = 128; s; s >>= 1) {
    if (threadIdx.x < s) sm[threadIdx.x] = fmaxf(sm[threadIdx.x], sm[threadIdx.x + s]);
    __syncthreads();
  }
  if (threadIdx.x == 0) sc[0] = sm[0];
}

__global__ __launch_bounds__(256)
void k_exp_partial(const float* __restrict__ A, const float* __restrict__ sc,
                   float* __restrict__ e, int N, float* __restrict__ part)
{
  __shared__ float sm[256];
  float mx = sc[0];
  float s = 0.0f;
  for (int i = blockIdx.x * 256 + threadIdx.x; i < N; i += gridDim.x * 256) {
    float v = __expf(A[i] - mx);
    e[i] = v;
    s += v;
  }
  sm[threadIdx.x] = s; __syncthreads();
  for (int st = 128; st; st >>= 1) {
    if (threadIdx.x < st) sm[threadIdx.x] += sm[threadIdx.x + st];
    __syncthreads();
  }
  if (threadIdx.x == 0) part[blockIdx.x] = sm[0];
}

__global__ __launch_bounds__(256)
void k_sum_final(const float* __restrict__ part, int G, float* __restrict__ sc)
{
  __shared__ float sm[256];
  float s = 0.0f;
  for (int i = threadIdx.x; i < G; i += 256) s += part[i];
  sm[threadIdx.x] = s; __syncthreads();
  for (int st = 128; st; st >>= 1) {
    if (threadIdx.x < st) sm[threadIdx.x] += sm[threadIdx.x + st];
    __syncthreads();
  }
  if (threadIdx.x == 0) sc[1] = sm[0];
}

// pooled[c] += sum_n e[n] * hpath[n, c]   (coalesced across threads)
__global__ __launch_bounds__(256)
void k_pool(const float* __restrict__ e, const float* __restrict__ hpath,
            float* __restrict__ pooled, int N)
{
  int t = threadIdx.x;
  float a0 = 0.0f, a1 = 0.0f;
  for (int n = blockIdx.x; n < N; n += gridDim.x) {
    float w = e[n];
    const float* hp = hpath + (size_t)n * CAT;
    a0 += w * hp[t];
    a1 += w * hp[t + 256];
  }
  atomicAdd(&pooled[t], a0);
  atomicAdd(&pooled[t + 256], a1);
}

// ---------------------------------------------------------------------------
// Head: rho -> wsi -> concat clin -> out1 -> out2 -> sigmoid/cumprod/argmax
// ---------------------------------------------------------------------------
__global__ __launch_bounds__(256)
void k_head(const float* __restrict__ pooled, const float* __restrict__ sc,
            const float* __restrict__ rho_w, const float* __restrict__ rho_b,
            const float* __restrict__ wsi_w, const float* __restrict__ wsi_b,
            const float* __restrict__ clin,
            const float* __restrict__ out1_w, const float* __restrict__ out1_b,
            const float* __restrict__ out2_w, const float* __restrict__ out2_b,
            float* __restrict__ out)
{
  __shared__ float hp[CAT];
  __shared__ float z1[CAT];
  __shared__ float full[WSI_H + CLIN];
  __shared__ float z2[CONCAT_H];
  int t = threadIdx.x;

  float inv = 1.0f / sc[1];
  for (int c = t; c < CAT; c += 256) hp[c] = pooled[c] * inv;
  __syncthreads();

  for (int c = t; c < CAT; c += 256) {                 // rho 512->512 relu
    float acc = rho_b[c];
    for (int k = 0; k < CAT; ++k) acc += hp[k] * rho_w[(size_t)k * CAT + c];
    z1[c] = fmaxf(acc, 0.0f);
  }
  __syncthreads();

  if (t < WSI_H) {                                     // wsi 512->256 relu
    float acc = wsi_b[t];
    for (int k = 0; k < CAT; ++k) acc += z1[k] * wsi_w[(size_t)k * WSI_H + t];
    full[t] = fmaxf(acc, 0.0f);
  }
  if (t < CLIN) full[WSI_H + t] = clin[t];
  __syncthreads();

  if (t < CONCAT_H) {                                  // out1 288->64 relu
    float acc = out1_b[t];
    for (int k = 0; k < WSI_H + CLIN; ++k) acc += full[k] * out1_w[k * CONCAT_H + t];
    z2[t] = fmaxf(acc, 0.0f);
  }
  __syncthreads();

  if (t == 0) {                                        // out2 64->4 + tail
    float logits[4], haz[4];
    for (int c = 0; c < 4; ++c) {
      float acc = out2_b[c];
      for (int k = 0; k < CONCAT_H; ++k) acc += z2[k] * out2_w[k * 4 + c];
      logits[c] = acc;
      haz[c] = 1.0f / (1.0f + __expf(-acc));
    }
    float S = 1.0f; int best = 0;
    for (int c = 0; c < 4; ++c) {
      out[c] = haz[c];
      S *= (1.0f - haz[c]);
      out[4 + c] = S;
      if (logits[c] > logits[best]) best = c;
    }
    out[8] = (float)best;
  }
}

// ---------------------------------------------------------------------------
// host-side orchestration
// ---------------------------------------------------------------------------
static void gemm(hipStream_t st, const void* A, int lda, int abf,
                 const unsigned short* Bt, const float* bias,
                 float* C, int ldc, unsigned short* Cbf,
                 int M, int N, int K, int act)
{
  dim3 grid((M + TM - 1) / TM, N / TN);
  if (abf) {
    if (act) k_gemm_wmma<1,1><<<grid, 256, 0, st>>>(A, lda, Bt, bias, C, ldc, Cbf, M, N, K);
    else     k_gemm_wmma<0,1><<<grid, 256, 0, st>>>(A, lda, Bt, bias, C, ldc, Cbf, M, N, K);
  } else {
    if (act) k_gemm_wmma<1,0><<<grid, 256, 0, st>>>(A, lda, Bt, bias, C, ldc, Cbf, M, N, K);
    else     k_gemm_wmma<0,0><<<grid, 256, 0, st>>>(A, lda, Bt, bias, C, ldc, Cbf, M, N, K);
  }
}

static void wprep(hipStream_t st, const void* W, unsigned short* Wt, int K, int N)
{
  k_wprep<<<(K * N + 255) / 256, 256, 0, st>>>((const float*)W, Wt, K, N);
}

extern "C" void kernel_launch(void* const* d_in, const int* in_sizes, int n_in,
                              void* d_out, int out_size, void* d_ws, size_t ws_size,
                              hipStream_t stream)
{
  (void)in_sizes; (void)n_in; (void)out_size; (void)ws_size;

  const float* x     = (const float*)d_in[IN_X];
  const int*   src   = (const int*)d_in[IN_EDGE];             // edge_index[0]
  const int*   dst   = ((const int*)d_in[IN_EDGE]) + N_EDGES; // edge_index[1]
  const float* clin  = (const float*)d_in[IN_CLIN];
  float* out = (float*)d_out;

  const size_t Nn = N_NODES;
  float* ws      = (float*)d_ws;
  float* xcat    = ws;                       // [N,512] fp32: h, h1, h2, h3
  float* hpath   = xcat  + Nn * 512;         // [N,512] fp32
  float* scratch = hpath + Nn * 512;         // [N,1024] (conv / attn union)
  float* mx      = scratch;                  // [N,128] (uint max bits)
  float* num     = scratch + Nn * 128;
  float* den     = scratch + Nn * 256;
  float* t1      = scratch + Nn * 384;       // [N,256]
  float* t2      = scratch + Nn * 640;       // [N,128]
  float* abuf    = scratch;                  // [N,512] (reuse, convs done)
  float* bbuf    = scratch + Nn * 512;       // [N,512]
  float* small   = scratch + Nn * 1024;
  float* Ascore  = small;                    // [N]
  float* evec    = small + Nn;               // [N]
  float* part    = small + 2 * Nn;           // [256]
  float* pooled  = part + 256;               // [512]
  float* sc      = pooled + 512;             // [2]: max, sum

  // bf16 region (16B aligned)
  unsigned short* us = (unsigned short*)
      (((uintptr_t)(sc + 2) + 15) & ~(uintptr_t)15);
  unsigned short* xcat_bf  = us;  us += Nn * 512;     // L2-resident A operand
  unsigned short* hpath_bf = us;  us += Nn * 512;
  unsigned short* node_bf  = us;  us += Nn * 128;
  unsigned short* t1_bf    = us;  us += Nn * 256;
  unsigned short* wt_fc    = us;  us += (size_t)D_IN * HID;
  unsigned short* wt_l1[3], *wt_l2[3];
  for (int l = 0; l < 3; ++l) { wt_l1[l] = us; us += HID * 2 * HID;
                                wt_l2[l] = us; us += 2 * HID * HID; }
  unsigned short* wt_phi = us;  us += (size_t)CAT * CAT;
  unsigned short* wt_a   = us;  us += (size_t)CAT * CAT;
  unsigned short* wt_b   = us;  us += (size_t)CAT * CAT;

  const dim3 blk(256);
  const int edge_grid = (N_EDGES + 1) / 2;
  const int ncGrid    = (N_NODES * HID + 255) / 256;
  const int rowGrid   = (N_NODES + 7) / 8;

  // ---- one-time weight conversion W[K,N] -> bf16 Wt[N,K] -------------------
  wprep(stream, d_in[P_FC_W], wt_fc, D_IN, HID);
  for (int l = 0; l < 3; ++l) {
    const int cp = P_CONV0 + l * 7;
    wprep(stream, d_in[cp + CV_LIN1_W], wt_l1[l], HID, 2 * HID);
    wprep(stream, d_in[cp + CV_LIN2_W], wt_l2[l], 2 * HID, HID);
  }
  wprep(stream, d_in[P_PHI_W],    wt_phi, CAT, CAT);
  wprep(stream, d_in[P_ATTN_A_W], wt_a,   CAT, CAT);
  wprep(stream, d_in[P_ATTN_B_W], wt_b,   CAT, CAT);

  // ---- fc: x[60000,1024] @ W -> xcat cols 0..127 (+bf16 shadow), relu ------
  gemm(stream, x, D_IN, 0, wt_fc, (const float*)d_in[P_FC_B],
       xcat, CAT, xcat_bf, N_NODES, HID, D_IN, 1);

  // ---- 3x GENConv ----------------------------------------------------------
  for (int l = 0; l < 3; ++l) {
    const int cp = P_CONV0 + l * 7;
    const float* hin = xcat + (size_t)l * HID;   // fp32 column slice, ld=512
    const float* tptr = (const float*)d_in[cp + CV_T];

    (void)hipMemsetAsync(mx,  0, Nn * HID * sizeof(float), stream);
    (void)hipMemsetAsync(num, 0, Nn * HID * sizeof(float), stream);
    (void)hipMemsetAsync(den, 0, Nn * HID * sizeof(float), stream);

    k_edge_max<<<edge_grid, blk, 0, stream>>>(src, dst, hin, CAT,
                                              (unsigned int*)mx, tptr, N_EDGES);
    k_edge_acc<<<edge_grid, blk, 0, stream>>>(src, dst, hin, CAT,
                                              (const unsigned int*)mx, num, den,
                                              tptr, N_EDGES);
    k_combine<<<ncGrid, blk, 0, stream>>>(num, den, hin, CAT, node_bf);

    // lin1: 128->256 (bf16 A)
    gemm(stream, node_bf, HID, 1, wt_l1[l], (const float*)d_in[cp + CV_LIN1_B],
         t1, 2 * HID, nullptr, N_NODES, 2 * HID, HID, 0);
    // relu(LN(t1)) -> t1_bf for lin2
    k_layernorm_relu<<<rowGrid, blk, 0, stream>>>(
        t1, 2 * HID, 2 * HID,
        (const float*)d_in[cp + CV_LN_G], (const float*)d_in[cp + CV_LN_B],
        nullptr, 0, nullptr, t1_bf, 2 * HID, N_NODES);

    if (l == 0) {
      // bare conv: lin2 straight into xcat cols 128..255 (+bf16 shadow)
      gemm(stream, t1_bf, 2 * HID, 1, wt_l2[l], (const float*)d_in[cp + CV_LIN2_B],
           xcat + HID, CAT, xcat_bf + HID, N_NODES, HID, 2 * HID, 0);
    } else {
      // lin2 -> t2, then h_{l+1} = h_l + relu(LN_norm(t2))
      gemm(stream, t1_bf, 2 * HID, 1, wt_l2[l], (const float*)d_in[cp + CV_LIN2_B],
           t2, HID, nullptr, N_NODES, HID, 2 * HID, 0);
      const float* ng = (const float*)d_in[(l == 1) ? P_NORM0_G : P_NORM1_G];
      const float* nb = (const float*)d_in[(l == 1) ? P_NORM0_B : P_NORM1_B];
      k_layernorm_relu<<<rowGrid, blk, 0, stream>>>(
          t2, HID, HID, ng, nb,
          hin, CAT,                                   // residual = conv input
          xcat + (size_t)(l + 1) * HID,
          xcat_bf + (size_t)(l + 1) * HID, CAT, N_NODES);
    }
  }

  // ---- attention MIL head over x_cat ---------------------------------------
  gemm(stream, xcat_bf, CAT, 1, wt_phi, (const float*)d_in[P_PHI_B],
       hpath, CAT, hpath_bf, N_NODES, CAT, CAT, 1);
  gemm(stream, hpath_bf, CAT, 1, wt_a, (const float*)d_in[P_ATTN_A_B],
       abuf, CAT, nullptr, N_NODES, CAT, CAT, 0);
  gemm(stream, hpath_bf, CAT, 1, wt_b, (const float*)d_in[P_ATTN_B_B],
       bbuf, CAT, nullptr, N_NODES, CAT, CAT, 0);

  k_attn_score<<<rowGrid, blk, 0, stream>>>(abuf, bbuf,
      (const float*)d_in[P_ATTN_C_W], (const float*)d_in[P_ATTN_C_B],
      Ascore, N_NODES);

  k_max_partial<<<256, blk, 0, stream>>>(Ascore, N_NODES, part);
  k_max_final<<<1, blk, 0, stream>>>(part, 256, sc);
  k_exp_partial<<<256, blk, 0, stream>>>(Ascore, sc, evec, N_NODES, part);
  k_sum_final<<<1, blk, 0, stream>>>(part, 256, sc);

  (void)hipMemsetAsync(pooled, 0, CAT * sizeof(float), stream);
  k_pool<<<240, blk, 0, stream>>>(evec, hpath, pooled, N_NODES);

  k_head<<<1, blk, 0, stream>>>(pooled, sc,
      (const float*)d_in[P_RHO_W], (const float*)d_in[P_RHO_B],
      (const float*)d_in[P_WSI_W], (const float*)d_in[P_WSI_B],
      clin,
      (const float*)d_in[P_OUT1_W], (const float*)d_in[P_OUT1_B],
      (const float*)d_in[P_OUT2_W], (const float*)d_in[P_OUT2_B],
      out);
}
